// Attention_Methods_45956150067908
// MI455X (gfx1250) — matrix operand order repaired and checked by reading, not compiled, not run on hardware
//
#include <hip/hip_runtime.h>
#include <hip/hip_bf16.h>
#include <math.h>

// Problem constants (from reference): B=16, L=4096, D=1024, A=256
#define BB 16
#define LL 4096
#define DD 1024
#define AA 256

typedef __attribute__((ext_vector_type(16))) _Float16 v16h;
typedef __attribute__((ext_vector_type(8)))  _Float16 v8h;
typedef __attribute__((ext_vector_type(8)))  float    v8f;

// ---------------------------------------------------------------------------
// K0: convert Wq (256x1024) and Wk (256x1024) fp32 -> f16 into ws (512x1024)
// rows 0..255 = Wq, rows 256..511 = Wk
// ---------------------------------------------------------------------------
__global__ void k_convert_w(const float* __restrict__ Wq,
                            const float* __restrict__ Wk,
                            _Float16* __restrict__ Wh) {
    int i = blockIdx.x * 256 + threadIdx.x;           // 0 .. 512*1024-1
    float v = (i < AA * DD) ? Wq[i] : Wk[i - AA * DD];
    Wh[i] = (_Float16)v;
}

// ---------------------------------------------------------------------------
// K1: fused projection GEMM (WMMA f16->f32) + LayerNorm(q), LayerNorm(k)
//     + score[b,l] = sum_a tanh(qn+kn) * score_w[a]
//
// Block: 256 threads = 8 waves. Block tile: 64 rows of x, all 512 cols.
// Wave (w = wm*4 + wc): rows wm*32 .. wm*32+31 (two 16-row groups rg=0,1),
// q cols [wc*64, +64) and the matching k cols (Wh row = 256 + a).
// acc[rg][t]: t=0..3 -> q tiles, t=4..7 -> k tiles.
// Each B fragment feeds TWO wmmas (rg=0,1) -> halves L2 weight traffic.
// ---------------------------------------------------------------------------
__global__ void k_proj_ln_score(const float* __restrict__ x,
                                const _Float16* __restrict__ Wh,
                                const float* __restrict__ gq_,
                                const float* __restrict__ bq_,
                                const float* __restrict__ gk_,
                                const float* __restrict__ bk_,
                                const float* __restrict__ sw_,
                                float* __restrict__ scores) {
    const int tid    = threadIdx.x;
    const int w      = tid >> 5;          // wave 0..7
    const int lane   = tid & 31;
    const int half   = lane >> 4;         // K-half for A/B frags, row-half for C
    const int n      = lane & 15;         // col-in-tile / row-in-tile(A)
    const int wave_m = w >> 2;            // 0..1  (32-row group)
    const int wave_c = w & 3;             // 0..3  (64-col group of q and of k)

    // global rows for the two A-fragments (A rows indexed by lane&15)
    const int rowbase = blockIdx.x * 64 + wave_m * 32;
    const float* xrow0 = x + (size_t)(rowbase + n)      * DD;
    const float* xrow1 = x + (size_t)(rowbase + 16 + n) * DD;

    // B row pointers for the 8 tiles (q tiles 0..3, k tiles 4..7)
    const _Float16* wrow[8];
#pragma unroll
    for (int t = 0; t < 4; ++t) {
        int aq = wave_c * 64 + t * 16 + n;            // a index 0..255
        wrow[t]     = Wh + (size_t)aq * DD;           // q weights
        wrow[t + 4] = Wh + (size_t)(AA + aq) * DD;    // k weights
    }

    v8f acc[2][8];
#pragma unroll
    for (int rg = 0; rg < 2; ++rg)
#pragma unroll
        for (int t = 0; t < 8; ++t)
            acc[rg][t] = (v8f){0.f,0.f,0.f,0.f,0.f,0.f,0.f,0.f};

    // ---- K loop: 1024 in steps of 32 --------------------------------------
    for (int k0 = 0; k0 < DD; k0 += 32) {
        // A fragments: 16-bit A 16x32 layout.
        // elements 0..7  = K = half*8 + 0..7 ; 8..15 = K = 16 + half*8 + 0..7
        v16h af[2];
        {
            const float* xr = xrow0 + k0 + (half << 3);
            float4 f0 = *(const float4*)(xr);
            float4 f1 = *(const float4*)(xr + 4);
            float4 f2 = *(const float4*)(xr + 16);
            float4 f3 = *(const float4*)(xr + 20);
            v16h a;
            a[0]=(_Float16)f0.x; a[1]=(_Float16)f0.y; a[2]=(_Float16)f0.z; a[3]=(_Float16)f0.w;
            a[4]=(_Float16)f1.x; a[5]=(_Float16)f1.y; a[6]=(_Float16)f1.z; a[7]=(_Float16)f1.w;
            a[8]=(_Float16)f2.x; a[9]=(_Float16)f2.y; a[10]=(_Float16)f2.z; a[11]=(_Float16)f2.w;
            a[12]=(_Float16)f3.x; a[13]=(_Float16)f3.y; a[14]=(_Float16)f3.z; a[15]=(_Float16)f3.w;
            af[0] = a;
        }
        {
            const float* xr = xrow1 + k0 + (half << 3);
            float4 f0 = *(const float4*)(xr);
            float4 f1 = *(const float4*)(xr + 4);
            float4 f2 = *(const float4*)(xr + 16);
            float4 f3 = *(const float4*)(xr + 20);
            v16h a;
            a[0]=(_Float16)f0.x; a[1]=(_Float16)f0.y; a[2]=(_Float16)f0.z; a[3]=(_Float16)f0.w;
            a[4]=(_Float16)f1.x; a[5]=(_Float16)f1.y; a[6]=(_Float16)f1.z; a[7]=(_Float16)f1.w;
            a[8]=(_Float16)f2.x; a[9]=(_Float16)f2.y; a[10]=(_Float16)f2.z; a[11]=(_Float16)f2.w;
            a[12]=(_Float16)f3.x; a[13]=(_Float16)f3.y; a[14]=(_Float16)f3.z; a[15]=(_Float16)f3.w;
            af[1] = a;
        }

#pragma unroll
        for (int t = 0; t < 8; ++t) {
            // B fragment: elements 0..7 = K = half*8 + 0..7 ; 8..15 = K = 16 + half*8 + 0..7
            const _Float16* wr = wrow[t] + k0 + (half << 3);
            v8h b0 = *(const v8h*)(wr);
            v8h b1 = *(const v8h*)(wr + 16);
            v16h bf;
#pragma unroll
            for (int i = 0; i < 8; ++i) { bf[i] = b0[i]; bf[8 + i] = b1[i]; }
            acc[0][t] = __builtin_amdgcn_wmma_f32_16x16x32_f16(
                false, af[0], false, bf, (short)0, acc[0][t], false, false);
            acc[1][t] = __builtin_amdgcn_wmma_f32_16x16x32_f16(
                false, af[1], false, bf, (short)0, acc[1][t], false, false);
        }
    }

    // ---- LayerNorm stats ---------------------------------------------------
    // C layout: lane half 0 -> rows j=0..7, half 1 -> rows 8..15 (per rg)
    float qs1[2][8], qs2[2][8], ks1[2][8], ks2[2][8];
#pragma unroll
    for (int rg = 0; rg < 2; ++rg) {
#pragma unroll
        for (int j = 0; j < 8; ++j) {
            float s1 = 0.f, s2 = 0.f;
#pragma unroll
            for (int t = 0; t < 4; ++t) { float v = acc[rg][t][j]; s1 += v; s2 += v * v; }
            qs1[rg][j] = s1; qs2[rg][j] = s2;
            s1 = 0.f; s2 = 0.f;
#pragma unroll
            for (int t = 4; t < 8; ++t) { float v = acc[rg][t][j]; s1 += v; s2 += v * v; }
            ks1[rg][j] = s1; ks2[rg][j] = s2;
        }
    }
    // reduce over the 16 lanes that share the same row-half
#pragma unroll
    for (int m = 1; m < 16; m <<= 1) {
#pragma unroll
        for (int rg = 0; rg < 2; ++rg)
#pragma unroll
            for (int j = 0; j < 8; ++j) {
                qs1[rg][j] += __shfl_xor(qs1[rg][j], m, 32);
                qs2[rg][j] += __shfl_xor(qs2[rg][j], m, 32);
                ks1[rg][j] += __shfl_xor(ks1[rg][j], m, 32);
                ks2[rg][j] += __shfl_xor(ks2[rg][j], m, 32);
            }
    }

    __shared__ float s1q[8][32], s2q[8][32], s1k[8][32], s2k[8][32], sc[8][32];
    if (n == 0) {  // lanes 0 and 16 publish their rows
#pragma unroll
        for (int rg = 0; rg < 2; ++rg)
#pragma unroll
            for (int j = 0; j < 8; ++j) {
                int r32 = rg * 16 + (half << 3) + j;
                s1q[w][r32] = qs1[rg][j]; s2q[w][r32] = qs2[rg][j];
                s1k[w][r32] = ks1[rg][j]; s2k[w][r32] = ks2[rg][j];
            }
    }
    __syncthreads();

    // per-row mean / rstd over the full 256 channels (combine 4 col groups)
    float mq[2][8], rq[2][8], mk[2][8], rk[2][8];
    const int wbase = wave_m * 4;
#pragma unroll
    for (int rg = 0; rg < 2; ++rg) {
#pragma unroll
        for (int j = 0; j < 8; ++j) {
            int r32 = rg * 16 + (half << 3) + j;
            float a1 = 0.f, a2 = 0.f, c1 = 0.f, c2 = 0.f;
#pragma unroll
            for (int c = 0; c < 4; ++c) {
                a1 += s1q[wbase + c][r32]; a2 += s2q[wbase + c][r32];
                c1 += s1k[wbase + c][r32]; c2 += s2k[wbase + c][r32];
            }
            float m1 = a1 * (1.0f / AA);
            mq[rg][j] = m1;
            rq[rg][j] = rsqrtf(fmaxf(a2 * (1.0f / AA) - m1 * m1, 0.f) + 1e-6f);
            float m2 = c1 * (1.0f / AA);
            mk[rg][j] = m2;
            rk[rg][j] = rsqrtf(fmaxf(c2 * (1.0f / AA) - m2 * m2, 0.f) + 1e-6f);
        }
    }

    // ---- fused LN + tanh + score dot (q and k live in the same lane) ------
    float p[2][8];
#pragma unroll
    for (int rg = 0; rg < 2; ++rg)
#pragma unroll
        for (int j = 0; j < 8; ++j) p[rg][j] = 0.f;
#pragma unroll
    for (int t = 0; t < 4; ++t) {
        int a = wave_c * 64 + t * 16 + n;
        float ga = gq_[a], ba = bq_[a], gc = gk_[a], bc = bk_[a], swv = sw_[a];
#pragma unroll
        for (int rg = 0; rg < 2; ++rg)
#pragma unroll
            for (int j = 0; j < 8; ++j) {
                float qn = (acc[rg][t][j]     - mq[rg][j]) * rq[rg][j] * ga + ba;
                float kn = (acc[rg][t + 4][j] - mk[rg][j]) * rk[rg][j] * gc + bc;
                p[rg][j] += tanhf(qn + kn) * swv;
            }
    }
#pragma unroll
    for (int m = 1; m < 16; m <<= 1)
#pragma unroll
        for (int rg = 0; rg < 2; ++rg)
#pragma unroll
            for (int j = 0; j < 8; ++j) p[rg][j] += __shfl_xor(p[rg][j], m, 32);

    if (n == 0) {
#pragma unroll
        for (int rg = 0; rg < 2; ++rg)
#pragma unroll
            for (int j = 0; j < 8; ++j)
                sc[w][rg * 16 + (half << 3) + j] = p[rg][j];
    }
    __syncthreads();

    if (tid < 64) {  // one thread per block row writes the score
        int r   = tid;
        int wm  = r >> 5, r32 = r & 31;
        float s = 0.f;
#pragma unroll
        for (int c = 0; c < 4; ++c) s += sc[wm * 4 + c][r32];
        scores[blockIdx.x * 64 + r] = s;
    }
}

// ---------------------------------------------------------------------------
// K2: masked softmax over L per batch. One block per batch row.
// ---------------------------------------------------------------------------
__global__ void k_softmax(const float* __restrict__ scores,
                          const unsigned char* __restrict__ mask,
                          float* __restrict__ att) {
    const int b = blockIdx.x, tid = threadIdx.x;
    __shared__ float red[256];
    float vals[16];
    float lmax = -INFINITY;
#pragma unroll
    for (int i = 0; i < 16; ++i) {
        int l = tid + i * 256;
        float s = scores[b * LL + l];
        if (mask[b * LL + l]) s = -INFINITY;
        vals[i] = s;
        lmax = fmaxf(lmax, s);
    }
    red[tid] = lmax; __syncthreads();
    for (int s = 128; s > 0; s >>= 1) {
        if (tid < s) red[tid] = fmaxf(red[tid], red[tid + s]);
        __syncthreads();
    }
    float m = red[0]; __syncthreads();
    float lsum = 0.f;
#pragma unroll
    for (int i = 0; i < 16; ++i) { float e = expf(vals[i] - m); vals[i] = e; lsum += e; }
    red[tid] = lsum; __syncthreads();
    for (int s = 128; s > 0; s >>= 1) {
        if (tid < s) red[tid] += red[tid + s];
        __syncthreads();
    }
    float inv = 1.0f / red[0];
#pragma unroll
    for (int i = 0; i < 16; ++i) att[b * LL + tid + i * 256] = vals[i] * inv;
}

// ---------------------------------------------------------------------------
// K3a: partial att·x.  grid (B, 8): each block handles 512 L-positions,
// 256 threads x float4 covers all 1024 channels.
// ---------------------------------------------------------------------------
__global__ void k_av_partial(const float* __restrict__ x,
                             const float* __restrict__ att,
                             float* __restrict__ part) {
    const int b = blockIdx.x, lc = blockIdx.y;
    const int d4 = threadIdx.x * 4;
    const float* xb = x + ((size_t)b * LL + (size_t)lc * 512) * DD + d4;
    const float* ab = att + b * LL + lc * 512;
    float ax = 0.f, ay = 0.f, az = 0.f, aw = 0.f;
    for (int l = 0; l < 512; ++l) {
        __builtin_prefetch(xb + (size_t)(l + 4) * DD, 0, 1);
        float a  = ab[l];
        float4 v = *(const float4*)(xb + (size_t)l * DD);
        ax += a * v.x; ay += a * v.y; az += a * v.z; aw += a * v.w;
    }
    float4 o; o.x = ax; o.y = ay; o.z = az; o.w = aw;
    *(float4*)(part + ((size_t)(b * 8 + lc)) * DD + d4) = o;
}

// ---------------------------------------------------------------------------
// K3b: deterministic reduction of the 8 partials -> attention_result [B, D]
// ---------------------------------------------------------------------------
__global__ void k_av_reduce(const float* __restrict__ part,
                            float* __restrict__ result) {
    const int b = blockIdx.x;
    const int d4 = threadIdx.x * 4;
    float ax = 0.f, ay = 0.f, az = 0.f, aw = 0.f;
#pragma unroll
    for (int lc = 0; lc < 8; ++lc) {
        float4 v = *(const float4*)(part + ((size_t)(b * 8 + lc)) * DD + d4);
        ax += v.x; ay += v.y; az += v.z; aw += v.w;
    }
    float4 o; o.x = ax; o.y = ay; o.z = az; o.w = aw;
    *(float4*)(result + (size_t)b * DD + d4) = o;
}

// ---------------------------------------------------------------------------
extern "C" void kernel_launch(void* const* d_in, const int* in_sizes, int n_in,
                              void* d_out, int out_size, void* d_ws, size_t ws_size,
                              hipStream_t stream) {
    const float*         x    = (const float*)d_in[0];
    const unsigned char* mask = (const unsigned char*)d_in[1];
    const float*         Wq   = (const float*)d_in[2];
    const float*         Wk   = (const float*)d_in[3];
    const float*         gq   = (const float*)d_in[4];
    const float*         bq   = (const float*)d_in[5];
    const float*         gk   = (const float*)d_in[6];
    const float*         bk   = (const float*)d_in[7];
    const float*         sw   = (const float*)d_in[8];

    float* out    = (float*)d_out;
    float* result = out;              // [B, D]   = 16384 floats
    float* att    = out + BB * DD;    // [B, 1, L] = 65536 floats

    char* wsb = (char*)d_ws;
    _Float16* Wh     = (_Float16*)wsb;                                   // 1 MB
    float*    scores = (float*)(wsb + (size_t)512 * 1024 * 2);           // 256 KB
    float*    part   = (float*)(wsb + (size_t)512 * 1024 * 2
                                     + (size_t)BB * LL * 4);             // 512 KB

    k_convert_w    <<<2048, 256, 0, stream>>>(Wq, Wk, Wh);
    k_proj_ln_score<<<(BB * LL) / 64, 256, 0, stream>>>(x, Wh, gq, bq, gk, bk, sw, scores);
    k_softmax      <<<BB, 256, 0, stream>>>(scores, mask, att);
    k_av_partial   <<<dim3(BB, 8), 256, 0, stream>>>(x, att, part);
    k_av_reduce    <<<BB, 256, 0, stream>>>(part, result);
}
